// Refinement_46943992545819
// MI455X (gfx1250) — matrix operand order repaired and checked by reading, not compiled, not run on hardware
//
#include <hip/hip_runtime.h>

typedef __attribute__((ext_vector_type(16))) _Float16 v16h;
typedef __attribute__((ext_vector_type(8)))  float    v8f;

#define EMBED 32
#define SEQL  512
#define LAMBDA_SOFTMAX 4.0f
#define LEAKY_SLOPE    0.1f
#define NWAVE 8
// Q is pre-scaled by lambda*log2(e) so softmax runs in base-2 domain
// (leaky relu is positively homogeneous, so the scale commutes with it).
#define QSCALE (LAMBDA_SOFTMAX * 1.4426950408889634f)

// ---- WMMA fragment helpers (A/B use contiguous k = 16*grp + s, not the ISA table order) ----

// A-matrix (16x32 f16) K-index for vgpr-pair i (0..7), lane group grp (0/1):
// lanes 0-15 hold K {0..15}, lanes 16-31 hold K {16..31}.
__device__ __forceinline__ int a_k0(int i, int grp) {
  return 2 * i + 16 * grp;
}

__device__ __forceinline__ v8f wmma_f16(v16h a, v16h b, v8f c) {
  return __builtin_amdgcn_wmma_f32_16x16x32_f16(
      /*neg_a=*/false, a, /*neg_b=*/false, b,
      /*c_mod=*/(short)0, c, /*reuse_a=*/false, /*reuse_b=*/false);
}

// B-matrix (32x16) fragment from a row-major f32 weight W[32][32], computing
// X @ W^T: B[k][n] = W[n0+n][k] -> 16 contiguous floats at W[(n0+col)*32+grp*16].
__device__ __forceinline__ v16h load_wfragB(const float* __restrict__ W,
                                            int n0, int col, int grp) {
  const float* p = W + (n0 + col) * EMBED + grp * 16;
  v16h r;
#pragma unroll
  for (int j = 0; j < 16; ++j) r[j] = (_Float16)p[j];
  return r;
}

// A-fragment from per-wave LDS staging: row-major 16 x `ld` f16, K origin kbase.
__device__ __forceinline__ v16h load_afrag_stage(const _Float16* st, int ld,
                                                 int kbase, int col, int grp) {
  v16h r;
#pragma unroll
  for (int i = 0; i < 8; ++i) {
    int k = kbase + a_k0(i, grp);
    r[2 * i]     = st[col * ld + k];
    r[2 * i + 1] = st[col * ld + k + 1];
  }
  return r;
}

// Store a 16x16 f32 D-fragment into staging (row-major, leading dim ld) as f16
// at column origin n0. D layout: element [v + 8*grp][col] in vgpr v.
__device__ __forceinline__ void store_dtile_stage(_Float16* st, int ld, v8f t,
                                                  int n0, int col, int grp) {
#pragma unroll
  for (int v = 0; v < 8; ++v)
    st[(v + 8 * grp) * ld + n0 + col] = (_Float16)t[v];
}

// Row reductions across the 16 lanes of each half-wave (xor masks 1..8 never
// cross bit 4, so lane groups 0-15 / 16-31 reduce independently).
__device__ __forceinline__ float rmax16(float x) {
#pragma unroll
  for (int off = 8; off >= 1; off >>= 1) x = fmaxf(x, __shfl_xor(x, off, 32));
  return x;
}
__device__ __forceinline__ float rsum16(float x) {
#pragma unroll
  for (int off = 8; off >= 1; off >>= 1) x += __shfl_xor(x, off, 32);
  return x;
}

__global__ __launch_bounds__(256) void fused_attn_refine(
    const float* __restrict__ rgn, const float* __restrict__ wrd,
    const float* __restrict__ Wscale, const float* __restrict__ bscale,
    const float* __restrict__ W1, const float* __restrict__ b1,
    const float* __restrict__ W2, const float* __restrict__ b2,
    float* __restrict__ out) {
  __shared__ _Float16 sW [SEQL * EMBED];        // wrd f16 [s][d]  (K^T B-frags)
  __shared__ _Float16 sWT[EMBED * SEQL];        // wrd f16 [d][s]  (V B-frags)
  __shared__ float    sQm[SEQL];                // per-token mean of wrd
  __shared__ _Float16 stage[NWAVE][16 * 64];    // per-wave D->A relayout buffer

  const int b    = blockIdx.x;
  const int tid  = threadIdx.x;
  const int lane = tid & 31;
  const int wave = tid >> 5;
  const int col  = lane & 15;   // N / column within fragment
  const int grp  = lane >> 4;   // lane half

  const float* wrdB = wrd + (size_t)b * SEQL * EMBED;
  const float* rgnB = rgn + (size_t)b * SEQL * EMBED;
  float*       outB = out + (size_t)b * SEQL * EMBED;

  // ---- cooperative fill of LDS (f16 copies + per-row means) ----
  for (int i = tid; i < SEQL * EMBED; i += 256) {
    float v = wrdB[i];
    _Float16 h = (_Float16)v;
    int s = i >> 5, d = i & 31;
    sW[i] = h;
    sWT[d * SEQL + s] = h;
  }
  for (int r = tid; r < SEQL; r += 256) {
    float acc = 0.f;
#pragma unroll 8
    for (int d = 0; d < EMBED; ++d) acc += wrdB[r * EMBED + d];
    sQm[r] = acc * (1.0f / EMBED);
  }
  __syncthreads();

  const v8f zc = {};

  // Each wave owns q-tiles of 16 rows: 32 tiles / 8 waves = 4 per wave.
  for (int qt = wave; qt < SEQL / 16; qt += NWAVE) {
    const int qbase = qt * 16;

    // Q A-fragment straight from global (f32 -> f16), pre-scaled by lambda*log2e
    v16h aq;
    {
      const float* qsrc = rgnB + (qbase + col) * EMBED;
#pragma unroll
      for (int i = 0; i < 8; ++i) {
        int k = a_k0(i, grp);
        aq[2 * i]     = (_Float16)(qsrc[k] * QSCALE);
        aq[2 * i + 1] = (_Float16)(qsrc[k + 1] * QSCALE);
      }
    }

    float mrow[8], lrow[8];
    v8f O0 = {}, O1 = {};
#pragma unroll
    for (int v = 0; v < 8; ++v) { mrow[v] = -1e30f; lrow[v] = 0.f; }

    // ---- flash loop over source positions, 64 at a time ----
    for (int sc = 0; sc < SEQL; sc += 64) {
      // S = Q x K^T, four 16-col s-subtiles (K-dim = EMBED = 32, one WMMA each)
      v8f z[4];
#pragma unroll
      for (int j = 0; j < 4; ++j) {
        v16h bk = *(const v16h*)(sW + (sc + 16 * j + col) * EMBED + grp * 16);
        z[j] = wmma_f16(aq, bk, zc);
      }

#pragma unroll
      for (int v = 0; v < 8; ++v) {
        // logits (already *lambda*log2e): leaky(s) = max(s, 0.1*s)
        float s0 = fmaxf(z[0][v], LEAKY_SLOPE * z[0][v]);
        float s1 = fmaxf(z[1][v], LEAKY_SLOPE * z[1][v]);
        float s2 = fmaxf(z[2][v], LEAKY_SLOPE * z[2][v]);
        float s3 = fmaxf(z[3][v], LEAKY_SLOPE * z[3][v]);
        float t  = rmax16(fmaxf(fmaxf(s0, s1), fmaxf(s2, s3)));
        float nm = fmaxf(mrow[v], t);
        float es = exp2f(mrow[v] - nm);
        mrow[v]  = nm;
        float e0 = exp2f(s0 - nm);
        float e1 = exp2f(s1 - nm);
        float e2 = exp2f(s2 - nm);
        float e3 = exp2f(s3 - nm);
        z[0][v] = e0; z[1][v] = e1; z[2][v] = e2; z[3][v] = e3;
        lrow[v] = lrow[v] * es + rsum16((e0 + e1) + (e2 + e3));
        O0[v] *= es; O1[v] *= es;
      }

      // P (16x64, D layout) -> two A-fragments via per-wave LDS staging
      // (LDS ops are in-order within a wave; no barrier needed).
#pragma unroll
      for (int j = 0; j < 4; ++j)
        store_dtile_stage(stage[wave], 64, z[j], 16 * j, col, grp);
      v16h pa0 = load_afrag_stage(stage[wave], 64, 0, col, grp);
      v16h pa1 = load_afrag_stage(stage[wave], 64, 32, col, grp);

      // V B-fragments: B[k][n] = wrd[sc + h*32 + k'][d0+n], contiguous in sWT
#pragma unroll
      for (int h = 0; h < 2; ++h) {
        v16h pa = (h == 0) ? pa0 : pa1;
        v16h vb0 = *(const v16h*)(sWT + col * SEQL        + sc + 32 * h + grp * 16);
        v16h vb1 = *(const v16h*)(sWT + (16 + col) * SEQL + sc + 32 * h + grp * 16);
        O0 = wmma_f16(pa, vb0, O0);
        O1 = wmma_f16(pa, vb1, O1);
      }
    }

    // ---- normalize -> wei_ctx, then the per-token MLP (all WMMA) ----
#pragma unroll
    for (int v = 0; v < 8; ++v) {
      float iv = 1.0f / lrow[v];
      O0[v] *= iv; O1[v] *= iv;
    }
    store_dtile_stage(stage[wave], EMBED, O0, 0,  col, grp);
    store_dtile_stage(stage[wave], EMBED, O1, 16, col, grp);
    v16h ax = load_afrag_stage(stage[wave], EMBED, 0, col, grp);

    float qmv[8];
#pragma unroll
    for (int v = 0; v < 8; ++v) qmv[v] = sQm[qbase + v + 8 * grp];

    // scaling = tanh(ctx @ Wscale^T + b_scale); x = qm * scaling
    v16h wb0 = load_wfragB(Wscale, 0, col, grp);
    v16h wb1 = load_wfragB(Wscale, 16, col, grp);
    v8f t0 = wmma_f16(ax, wb0, zc);
    v8f t1 = wmma_f16(ax, wb1, zc);
    float ba = bscale[col], bb = bscale[16 + col];
#pragma unroll
    for (int v = 0; v < 8; ++v) {
      t0[v] = tanhf(t0[v] + ba) * qmv[v];
      t1[v] = tanhf(t1[v] + bb) * qmv[v];
    }
    store_dtile_stage(stage[wave], EMBED, t0, 0,  col, grp);
    store_dtile_stage(stage[wave], EMBED, t1, 16, col, grp);
    v16h axx = load_afrag_stage(stage[wave], EMBED, 0, col, grp);

    // h = relu(x @ W1^T + b1)
    wb0 = load_wfragB(W1, 0, col, grp);
    wb1 = load_wfragB(W1, 16, col, grp);
    t0 = wmma_f16(axx, wb0, zc);
    t1 = wmma_f16(axx, wb1, zc);
    ba = b1[col]; bb = b1[16 + col];
#pragma unroll
    for (int v = 0; v < 8; ++v) {
      t0[v] = fmaxf(t0[v] + ba, 0.f);
      t1[v] = fmaxf(t1[v] + bb, 0.f);
    }
    store_dtile_stage(stage[wave], EMBED, t0, 0,  col, grp);
    store_dtile_stage(stage[wave], EMBED, t1, 16, col, grp);
    v16h ah = load_afrag_stage(stage[wave], EMBED, 0, col, grp);

    // out = h @ W2^T + b2 + qm
    wb0 = load_wfragB(W2, 0, col, grp);
    wb1 = load_wfragB(W2, 16, col, grp);
    t0 = wmma_f16(ah, wb0, zc);
    t1 = wmma_f16(ah, wb1, zc);
    ba = b2[col]; bb = b2[16 + col];
    float* orow = outB + (size_t)qbase * EMBED;
#pragma unroll
    for (int v = 0; v < 8; ++v) {
      orow[(v + 8 * grp) * EMBED + col]      = t0[v] + ba + qmv[v];
      orow[(v + 8 * grp) * EMBED + 16 + col] = t1[v] + bb + qmv[v];
    }
  }
}

extern "C" void kernel_launch(void* const* d_in, const int* in_sizes, int n_in,
                              void* d_out, int out_size, void* d_ws, size_t ws_size,
                              hipStream_t stream) {
  (void)n_in; (void)out_size; (void)d_ws; (void)ws_size;
  const float* rgn    = (const float*)d_in[0];
  const float* wrd    = (const float*)d_in[1];
  const float* Wscale = (const float*)d_in[2];
  const float* bscale = (const float*)d_in[3];
  const float* W1     = (const float*)d_in[4];
  const float* b1     = (const float*)d_in[5];
  const float* W2     = (const float*)d_in[6];
  const float* b2     = (const float*)d_in[7];
  float* out = (float*)d_out;

  const int B = in_sizes[0] / (SEQL * EMBED);  // 256
  dim3 grid(B), block(256);
  fused_attn_refine<<<grid, block, 0, stream>>>(rgn, wrd, Wscale, bscale,
                                                W1, b1, W2, b2, out);
}
